// SharedMultiBandEncoder_8899172237597
// MI455X (gfx1250) — compile-verified, matrix operand order checked by reading, not run on hardware
//
#include <hip/hip_runtime.h>
#include <hip/hip_bf16.h>

// ---------------------------------------------------------------------------
// Problem constants (match reference)
// ---------------------------------------------------------------------------
#define NN     50000
#define EE     800000
#define ETOT   850000          // EE + NN self loops
#define NEGSL  0.2f
#define BNEPS  1e-5f

// ---------------------------------------------------------------------------
// WMMA types (CDNA5 / gfx1250, wave32)
// ---------------------------------------------------------------------------
typedef __attribute__((ext_vector_type(16))) __bf16 v16bf;
typedef __attribute__((ext_vector_type(8)))  float  v8f;

// ---------------------------------------------------------------------------
// helpers
// ---------------------------------------------------------------------------
__device__ __forceinline__ unsigned short f32_to_bf16_rne(float f) {
    unsigned u = __float_as_uint(f);
    u += 0x7fffu + ((u >> 16) & 1u);     // round-to-nearest-even
    return (unsigned short)(u >> 16);
}
// monotone encoding of float for atomicMax on unsigned
__device__ __forceinline__ unsigned fenc(float f) {
    unsigned u = __float_as_uint(f);
    return (u & 0x80000000u) ? ~u : (u | 0x80000000u);
}
__device__ __forceinline__ float fdec(unsigned e) {
    unsigned u = (e & 0x80000000u) ? (e & 0x7fffffffu) : ~e;
    return __uint_as_float(u);
}
#define FENC_NEG_INF 0x007FFFFFu   // fenc(-inf)

// ---------------------------------------------------------------------------
// fill kernels (grid-stride)
// ---------------------------------------------------------------------------
__global__ void k_fill_f32(float* __restrict__ p, float v, int cnt) {
    for (int i = blockIdx.x * blockDim.x + threadIdx.x; i < cnt;
         i += gridDim.x * blockDim.x) p[i] = v;
}
__global__ void k_fill_u32(unsigned* __restrict__ p, unsigned v, int cnt) {
    for (int i = blockIdx.x * blockDim.x + threadIdx.x; i < cnt;
         i += gridDim.x * blockDim.x) p[i] = v;
}

// ---------------------------------------------------------------------------
// f32 -> bf16 conversion (activations)
// ---------------------------------------------------------------------------
__global__ void k_f32_to_bf16(const float* __restrict__ X,
                              unsigned short* __restrict__ A, int cnt) {
    for (int i = blockIdx.x * blockDim.x + threadIdx.x; i < cnt;
         i += gridDim.x * blockDim.x) A[i] = f32_to_bf16_rne(X[i]);
}

// W [Fi,Fo] f32 -> Wt [Fo,Fi] bf16 (so B-fragment loads are contiguous)
__global__ void k_w_to_bf16_t(const float* __restrict__ W,
                              unsigned short* __restrict__ Wt, int Fi, int Fo) {
    int i = blockIdx.x * blockDim.x + threadIdx.x;
    if (i >= Fi * Fo) return;
    int fi = i / Fo, fo = i - fi * Fo;
    Wt[(size_t)fo * Fi + fi] = f32_to_bf16_rne(W[i]);
}

// ---------------------------------------------------------------------------
// bf16 WMMA GEMM: H[N,Fo] = A[N,Fi] * W[Fi,Fo]  (Wt = W^T, bf16)
// 4 waves / block; each wave computes one 16x16 tile of H.
// grid = (N/16, ceil(Fo/64))
// ---------------------------------------------------------------------------
__global__ __launch_bounds__(128)
void k_gemm_bf16(const unsigned short* __restrict__ A,
                 const unsigned short* __restrict__ Wt,
                 float* __restrict__ H, int Fi, int Fo) {
    const int wave = threadIdx.x >> 5;
    const int lane = threadIdx.x & 31;
    const int colBase = blockIdx.y * 64 + wave * 16;
    if (colBase >= Fo) return;                       // uniform per wave

    const int m  = lane & 15;                        // row-in-tile (A), col (B/D)
    const int hi = lane >> 4;                        // half-lane group
    const int row = blockIdx.x * 16 + m;

    // ISA 16-bit A layout: lane holds K = hi*8 + [0..7] and hi*8 + 16 + [0..7]
    const unsigned short* pa = A  + (size_t)row            * Fi + hi * 8;
    const unsigned short* pb = Wt + (size_t)(colBase + m)  * Fi + hi * 8;

    v8f acc = {};
    for (int k0 = 0; k0 < Fi; k0 += 32) {
        union { v16bf v; uint4 q[2]; } fa, fb;
        fa.q[0] = *(const uint4*)(pa + k0);
        fa.q[1] = *(const uint4*)(pa + k0 + 16);     // +16 elements = K+16..K+23
        fb.q[0] = *(const uint4*)(pb + k0);
        fb.q[1] = *(const uint4*)(pb + k0 + 16);
        acc = __builtin_amdgcn_wmma_f32_16x16x32_bf16(
            false, fa.v, false, fb.v, (short)0, acc, false, false);
    }
    // D layout: VGPR r holds row (r + hi*8), col = lane&15
    float* ph = H + (size_t)(blockIdx.x * 16) * Fo + colBase;
    #pragma unroll
    for (int r = 0; r < 8; ++r)
        ph[(size_t)(r + hi * 8) * Fo + m] = acc[r];
}

// ---------------------------------------------------------------------------
// per-node attention dots: ls[n]=h[n].a_src, ld[n]=h[n].a_dst (wave per node)
// ---------------------------------------------------------------------------
__global__ __launch_bounds__(128)
void k_dots(const float* __restrict__ H, const float* __restrict__ a_src,
            const float* __restrict__ a_dst, float* __restrict__ ls,
            float* __restrict__ ld, int n, int Fo) {
    int wid  = (blockIdx.x * blockDim.x + threadIdx.x) >> 5;
    int lane = threadIdx.x & 31;
    if (wid >= n) return;
    const float* hr = H + (size_t)wid * Fo;
    float s = 0.f, d = 0.f;
    for (int f = lane; f < Fo; f += 32) {
        float hv = hr[f];
        s += hv * a_src[f];
        d += hv * a_dst[f];
    }
    #pragma unroll
    for (int off = 16; off > 0; off >>= 1) {
        s += __shfl_down(s, off, 32);
        d += __shfl_down(d, off, 32);
    }
    if (lane == 0) { ls[wid] = s; ld[wid] = d; }
}

// ---------------------------------------------------------------------------
// edge pass 1: leaky-relu logits + segment max (atomicMax on encoded float)
// ---------------------------------------------------------------------------
__global__ void k_edge_logits(const int* __restrict__ src, const int* __restrict__ dst,
                              const float* __restrict__ ls, const float* __restrict__ ld,
                              float* __restrict__ ebuf, unsigned* __restrict__ menc) {
    int e = blockIdx.x * blockDim.x + threadIdx.x;
    if (e >= ETOT) return;
    int s = (e < EE) ? src[e] : (e - EE);
    int d = (e < EE) ? dst[e] : (e - EE);
    float l = ls[s] + ld[d];
    l = (l > 0.f) ? l : NEGSL * l;
    ebuf[e] = l;
    atomicMax(&menc[d], fenc(l));
}

// edge pass 2: exp(logit - max[dst]) + segment sum
__global__ void k_edge_exp(const int* __restrict__ dst, float* __restrict__ ebuf,
                           const unsigned* __restrict__ menc,
                           float* __restrict__ denom) {
    int e = blockIdx.x * blockDim.x + threadIdx.x;
    if (e >= ETOT) return;
    int d = (e < EE) ? dst[e] : (e - EE);
    float ex = expf(ebuf[e] - fdec(menc[d]));
    ebuf[e] = ex;
    atomicAdd(&denom[d], ex);
}

// edge pass 3: out[dst] += alpha * h[src]  (wave per edge, L2-resident atomics)
__global__ __launch_bounds__(256)
void k_edge_scatter(const int* __restrict__ src, const int* __restrict__ dst,
                    const float* __restrict__ ebuf, const float* __restrict__ denom,
                    const float* __restrict__ H, float* __restrict__ Out,
                    float* __restrict__ imp, int Fo, int doImp) {
    int wid  = (blockIdx.x * blockDim.x + threadIdx.x) >> 5;
    int lane = threadIdx.x & 31;
    if (wid >= ETOT) return;
    int s = (wid < EE) ? src[wid] : (wid - EE);
    int d = (wid < EE) ? dst[wid] : (wid - EE);
    float alpha = ebuf[wid] / denom[d];
    const float* hs = H + (size_t)s * Fo;
    float* od = Out + (size_t)d * Fo;
    for (int f = lane; f < Fo; f += 32)
        atomicAdd(&od[f], alpha * hs[f]);
    if (doImp && lane == 0 && s != d)
        atomicAdd(&imp[d], alpha);
}

// ---------------------------------------------------------------------------
// BatchNorm column stats: blockDim.x == Fo; gridDim.x row chunks
// ---------------------------------------------------------------------------
__global__ void k_bn_stats(const float* __restrict__ Out, float* __restrict__ sums,
                           float* __restrict__ sumsq, int n, int Fo) {
    int f = threadIdx.x;
    float s = 0.f, s2 = 0.f;
    for (int r = blockIdx.x; r < n; r += gridDim.x) {
        float v = Out[(size_t)r * Fo + f];
        s += v; s2 += v * v;
    }
    atomicAdd(&sums[f], s);
    atomicAdd(&sumsq[f], s2);
}

// bias b cancels exactly inside BN (constant shift), so it is omitted.
__global__ void k_bn_finalize(const float* __restrict__ sums, const float* __restrict__ sumsq,
                              const float* __restrict__ gamma, const float* __restrict__ beta,
                              float* __restrict__ scale, float* __restrict__ shift,
                              int n, int Fo) {
    int f = blockIdx.x * blockDim.x + threadIdx.x;
    if (f >= Fo) return;
    float mean = sums[f] / (float)n;
    float var  = sumsq[f] / (float)n - mean * mean;
    float sc   = gamma[f] * rsqrtf(var + BNEPS);
    scale[f] = sc;
    shift[f] = beta[f] - mean * sc;
}

// BN+ReLU apply: write bf16 activations for next layer; f32 to d_out on last layer
__global__ void k_bn_apply(const float* __restrict__ Out, const float* __restrict__ scale,
                           const float* __restrict__ shift, unsigned short* __restrict__ Abf,
                           float* __restrict__ Z, int n, int Fo) {
    int total = n * Fo;
    for (int i = blockIdx.x * blockDim.x + threadIdx.x; i < total;
         i += gridDim.x * blockDim.x) {
        int f = i % Fo;
        float v = fmaxf(Out[i] * scale[f] + shift[f], 0.f);
        Abf[i] = f32_to_bf16_rne(v);
        if (Z) Z[i] = v;
    }
}

// ---------------------------------------------------------------------------
// host orchestration
// ---------------------------------------------------------------------------
extern "C" void kernel_launch(void* const* d_in, const int* in_sizes, int n_in,
                              void* d_out, int out_size, void* d_ws, size_t ws_size,
                              hipStream_t stream) {
    (void)in_sizes; (void)n_in; (void)out_size; (void)ws_size;
    const int FiL[3] = {128, 256, 128};
    const int FoL[3] = {256, 128, 64};

    // ---- workspace carve (256B aligned bump allocator) ----
    char* ws = (char*)d_ws;
    size_t off = 0;
    auto alloc = [&](size_t bytes) -> void* {
        void* p = ws + off;
        off = (off + bytes + 255) & ~(size_t)255;
        return p;
    };
    unsigned short* Abf = (unsigned short*)alloc((size_t)NN * 256 * 2); // activations bf16
    unsigned short* Wt  = (unsigned short*)alloc((size_t)256 * 256 * 2);
    float* H     = (float*)alloc((size_t)NN * 256 * 4);
    float* Out   = (float*)alloc((size_t)NN * 256 * 4);
    float* ls    = (float*)alloc((size_t)NN * 4);
    float* ldv   = (float*)alloc((size_t)NN * 4);
    unsigned* me = (unsigned*)alloc((size_t)NN * 4);
    float* denom = (float*)alloc((size_t)NN * 4);
    float* ebuf  = (float*)alloc((size_t)ETOT * 4);
    float* sums  = (float*)alloc(256 * 4);
    float* sumsq = (float*)alloc(256 * 4);
    float* scale = (float*)alloc(256 * 4);
    float* shift = (float*)alloc(256 * 4);

    float* zout   = (float*)d_out;           // [3][N][64]
    float* impout = zout + (size_t)3 * NN * 64; // [3][N]

    const dim3 B256(256);

    for (int band = 0; band < 3; ++band) {
        const float* x    = (const float*)d_in[band * 20 + 0];
        const int*   eidx = (const int*)  d_in[band * 20 + 1];
        const int* srcp = eidx;          // row 0 of [2,E]
        const int* dstp = eidx + EE;     // row 1
        float* imp = impout + (size_t)band * NN;

        k_fill_f32<<<512, B256, 0, stream>>>(imp, 0.f, NN);
        k_f32_to_bf16<<<4096, B256, 0, stream>>>(x, Abf, NN * 128);

        for (int L = 0; L < 3; ++L) {
            // params: W, a_dst, a_src, b, beta, gamma (pytree-sorted keys)
            const float* W     = (const float*)d_in[band * 20 + 2 + L * 6 + 0];
            const float* a_dst = (const float*)d_in[band * 20 + 2 + L * 6 + 1];
            const float* a_src = (const float*)d_in[band * 20 + 2 + L * 6 + 2];
            // b (idx+3) cancels inside BatchNorm -> unused
            const float* beta  = (const float*)d_in[band * 20 + 2 + L * 6 + 4];
            const float* gamma = (const float*)d_in[band * 20 + 2 + L * 6 + 5];
            const int fi = FiL[L], fo = FoL[L];

            k_w_to_bf16_t<<<(fi * fo + 255) / 256, B256, 0, stream>>>(W, Wt, fi, fo);

            dim3 gg(NN / 16, (fo + 63) / 64);
            k_gemm_bf16<<<gg, 128, 0, stream>>>(Abf, Wt, H, fi, fo);

            k_dots<<<(NN + 3) / 4, 128, 0, stream>>>(H, a_src, a_dst, ls, ldv, NN, fo);

            k_fill_f32<<<4096, B256, 0, stream>>>(Out, 0.f, NN * fo);
            k_fill_u32<<<512, B256, 0, stream>>>(me, FENC_NEG_INF, NN);
            k_fill_f32<<<512, B256, 0, stream>>>(denom, 0.f, NN);

            k_edge_logits<<<(ETOT + 255) / 256, B256, 0, stream>>>(srcp, dstp, ls, ldv, ebuf, me);
            k_edge_exp<<<(ETOT + 255) / 256, B256, 0, stream>>>(dstp, ebuf, me, denom);
            k_edge_scatter<<<(ETOT + 7) / 8, B256, 0, stream>>>(
                srcp, dstp, ebuf, denom, H, Out, imp, fo, (L == 2) ? 1 : 0);

            k_fill_f32<<<4, B256, 0, stream>>>(sums, 0.f, fo);
            k_fill_f32<<<4, B256, 0, stream>>>(sumsq, 0.f, fo);
            k_bn_stats<<<512, fo, 0, stream>>>(Out, sums, sumsq, NN, fo);
            k_bn_finalize<<<1, 256, 0, stream>>>(sums, sumsq, gamma, beta, scale, shift, NN, fo);

            float* z = (L == 2) ? (zout + (size_t)band * NN * 64) : nullptr;
            k_bn_apply<<<4096, B256, 0, stream>>>(Out, scale, shift, Abf, z, NN, fo);
        }
    }
}
